// CapacityRouter_86406152061622
// MI455X (gfx1250) — matrix operand order repaired and checked by reading, not compiled, not run on hardware
//
#include <hip/hip_runtime.h>
#include <hip/hip_bf16.h>

// ---------------------------------------------------------------------------
// MoE capacity router for MI455X (gfx1250, wave32).
//  K0: one-time W fp32 -> bf16, transposed [e][k] (256 KB -> 128 KB, L2-hot).
//  K1: x@W logits via v_wmma_f32_16x16x32_bf16 (fp32 accum), softmax, top-2.
//      HBM-bound: reads x (1.07 GB) once -> ~46us floor @ 23.3 TB/s; bf16 WMMA
//      compute sits well under the floor. B tile staged with
//      global_load_async_to_lds_b128 (ASYNCcnt), A tile converted in-regs.
//  K2-K4: exact sequential-order capacity masking (segmented prefix count).
// ---------------------------------------------------------------------------

#define HIDDEN   1024
#define NEXP     64
#define CAPACITY 10240          // int(N*K/E * 1.25) = int(8192*1.25)
#define TPB_TOK  128            // tokens per block in K1
#define KSTEP    32
#define ASTRIDE  40             // padded bf16 row stride: 80B, 16B-aligned, no bank conflicts
#define LSTRIDE  65             // logits LDS row stride (floats), conflict-free
#define SCAN_CHUNK 4096         // assignments per scan block (256 thr * 16)

typedef __attribute__((ext_vector_type(16))) __bf16 v16bf;
typedef __attribute__((ext_vector_type(8)))  float  v8f;
typedef __attribute__((ext_vector_type(4)))  int    v4i;

union FragU { uint4 u[2]; v16bf v; };

#if __has_builtin(__builtin_amdgcn_global_load_async_to_lds_b128)
#define HAS_ASYNC_LDS 1
typedef __attribute__((address_space(1))) void* as1_vptr;
typedef __attribute__((address_space(3))) void* as3_vptr;
typedef __attribute__((address_space(1))) v4i*  as1_v4i;
typedef __attribute__((address_space(3))) v4i*  as3_v4i;
#else
#define HAS_ASYNC_LDS 0
#endif

__device__ __forceinline__ unsigned short f32_to_bf16_rne(float f) {
  unsigned int b = __float_as_uint(f);
  b += 0x7FFFu + ((b >> 16) & 1u);           // round to nearest even
  return (unsigned short)(b >> 16);
}
__device__ __forceinline__ unsigned int pack_bf16(float lo, float hi) {
  return (unsigned int)f32_to_bf16_rne(lo) | ((unsigned int)f32_to_bf16_rne(hi) << 16);
}

// ---------------------------------------------------------------------------
// Kernel 0: W [k][e] fp32 -> Wt [e][k] bf16 (one-time, 64K elements).
// ---------------------------------------------------------------------------
__global__ __launch_bounds__(256) void convert_W(
    const float* __restrict__ W, unsigned short* __restrict__ Wt)
{
  const int e = blockIdx.x;                       // 64 blocks, one expert each
  for (int k = threadIdx.x; k < HIDDEN; k += 256)
    Wt[(size_t)e * HIDDEN + k] = f32_to_bf16_rne(W[(size_t)k * NEXP + e]);
}

// ---------------------------------------------------------------------------
// Kernel 1: GEMM (bf16 WMMA, fp32 accum) + softmax + top-2 per token.
// Block: 256 threads = 8 waves; wave w owns tokens 16w..16w+15 x all 64 experts
// as 4 accumulators (16x16 tiles over expert groups of 16).
// ---------------------------------------------------------------------------
__global__ __launch_bounds__(256) void router_gemm_topk(
    const float* __restrict__ x, const unsigned short* __restrict__ Wt,
    float* __restrict__ out_idx,            // d_out[0 .. 2N)
    unsigned char* __restrict__ eid,        // ws: expert ids  [2N]
    float* __restrict__ pw)                 // ws: pre-mask topk weights [2N]
{
  __shared__ unsigned short As[TPB_TOK * ASTRIDE];   // 10.0 KB
  __shared__ unsigned short Bs[NEXP    * ASTRIDE];   //  5.0 KB
  __shared__ float          Ls[TPB_TOK * LSTRIDE];   // 32.5 KB

  const int tid  = threadIdx.x;
  const int wave = tid >> 5;
  const int lane = tid & 31;
  const int tok0 = blockIdx.x * TPB_TOK;

  v8f acc[4] = {v8f{}, v8f{}, v8f{}, v8f{}};

  // A-staging mapping: thread -> (token row, 16-float half of the 32-k chunk)
  const int ar = tid >> 1;
  const int ah = (tid & 1) * 16;
  const float* xrow = x + (size_t)(tok0 + ar) * HIDDEN + ah;
  // B-staging mapping: thread -> 16B slice of the 64x32 bf16 W tile
  const int ecopy = tid >> 2;                 // expert row 0..63
  const int kpart = (tid & 3) * 8;            // 8 bf16 = 16B per thread
  const unsigned short* wtrow = Wt + (size_t)ecopy * HIDDEN + kpart;
  unsigned short* bdst = &Bs[ecopy * ASTRIDE + kpart];   // 16B-aligned

  // Fragment addressing (ISA 7.12.2, wave32):
  //  A 16x32 bf16: lanes 0-15 = rows, elems 0-7 = K0..7, 8-15 = K16..23;
  //                lanes 16-31 hold K8..15 / K24..31  -> koff = (lane>>4)*8.
  //  B 32x16 bf16: lane = column, same K order as A (elems 0-7 = K koff..+7,
  //                8-15 = K 16+koff..) -> koff = (lane>>4)*8.
  const int arow  = wave * 16 + (lane & 15);
  const int akoff = (lane >> 4) * 8;
  const int bcol  = lane & 15;
  const int bkoff = (lane >> 4) * 8;

  for (int kb = 0; kb < HIDDEN; kb += KSTEP) {
    // ---- stage B: 64x32 bf16 tile, async global -> LDS (ASYNCcnt) ----
#if HAS_ASYNC_LDS
    __builtin_amdgcn_global_load_async_to_lds_b128(
        (as1_v4i)(as1_vptr)(void*)const_cast<unsigned short*>(wtrow + kb),
        (as3_v4i)(as3_vptr)(void*)bdst, 0, 0);
#else
    *(uint4*)bdst = *(const uint4*)(wtrow + kb);
#endif

    if (kb + KSTEP < HIDDEN)                   // WGP-scope prefetch of next A
      __builtin_prefetch((const void*)(xrow + kb + KSTEP), 0, 3);

    // ---- stage A: 128x32 fp32 -> bf16 (RNE) into LDS ----
    float4 f0 = *(const float4*)(xrow + kb + 0);
    float4 f1 = *(const float4*)(xrow + kb + 4);
    float4 f2 = *(const float4*)(xrow + kb + 8);
    float4 f3 = *(const float4*)(xrow + kb + 12);
    uint4 p0, p1;
    p0.x = pack_bf16(f0.x, f0.y); p0.y = pack_bf16(f0.z, f0.w);
    p0.z = pack_bf16(f1.x, f1.y); p0.w = pack_bf16(f1.z, f1.w);
    p1.x = pack_bf16(f2.x, f2.y); p1.y = pack_bf16(f2.z, f2.w);
    p1.z = pack_bf16(f3.x, f3.y); p1.w = pack_bf16(f3.z, f3.w);
    uint4* adst = (uint4*)&As[ar * ASTRIDE + ah];    // 80B row stride -> 16B aligned
    adst[0] = p0;
    adst[1] = p1;

#if HAS_ASYNC_LDS
#if __has_builtin(__builtin_amdgcn_s_wait_asynccnt)
    __builtin_amdgcn_s_wait_asynccnt(0);
#else
    asm volatile("s_wait_asynccnt 0" ::: "memory");
#endif
#endif
    __syncthreads();

    // ---- fragments + 4 WMMAs (one per 16-expert tile) ----
    FragU a;
    a.u[0] = *(const uint4*)&As[arow * ASTRIDE + akoff];       // K koff..koff+7
    a.u[1] = *(const uint4*)&As[arow * ASTRIDE + 16 + akoff];  // K 16+koff..
#pragma unroll
    for (int t = 0; t < 4; ++t) {
      FragU b;
      const int e = t * 16 + bcol;
      b.u[0] = *(const uint4*)&Bs[e * ASTRIDE + bkoff];
      b.u[1] = *(const uint4*)&Bs[e * ASTRIDE + bkoff + 16];
      acc[t] = __builtin_amdgcn_wmma_f32_16x16x32_bf16(
          false, a.v, false, b.v, (short)0, acc[t], false, false);
    }
    __syncthreads();
  }

  // ---- scatter logits to LDS (C layout: VGPR r -> M=r (lo lanes) / r+8) ----
#pragma unroll
  for (int t = 0; t < 4; ++t)
#pragma unroll
    for (int r = 0; r < 8; ++r) {
      const int trow = wave * 16 + r + ((lane >> 4) << 3);
      Ls[trow * LSTRIDE + t * 16 + (lane & 15)] = acc[t][r];
    }
  __syncthreads();

  // ---- per-token softmax + top-2 (threads 0..127) ----
  if (tid < TPB_TOK) {
    const float* lp = &Ls[tid * LSTRIDE];
    float m = lp[0];
    for (int e = 1; e < NEXP; ++e) m = fmaxf(m, lp[e]);
    float v1 = -1.0f, v2 = -1.0f;
    int   i1 = 0, i2 = 0;
    for (int e = 0; e < NEXP; ++e) {
      const float p = __expf(lp[e] - m);         // softmax numerator; monotone
      if (p > v1)      { v2 = v1; i2 = i1; v1 = p; i1 = e; }
      else if (p > v2) { v2 = p;  i2 = e; }
    }
    const float inv = 1.0f / (v1 + v2);          // sum cancels in the ratio
    const int gt = tok0 + tid;
    out_idx[2 * gt + 0] = (float)i1;
    out_idx[2 * gt + 1] = (float)i2;
    eid[2 * gt + 0] = (unsigned char)i1;
    eid[2 * gt + 1] = (unsigned char)i2;
    pw[2 * gt + 0] = v1 * inv;
    pw[2 * gt + 1] = v2 * inv;
  }
}

// ---------------------------------------------------------------------------
// Kernel 2: per-chunk expert histograms (4096 assignments / block).
// ---------------------------------------------------------------------------
__global__ __launch_bounds__(256) void hist_kernel(
    const unsigned char* __restrict__ eid, int* __restrict__ hist)
{
  __shared__ int cnt[NEXP];
  const int tid = threadIdx.x;
  if (tid < NEXP) cnt[tid] = 0;
  __syncthreads();
  const unsigned char* p = eid + (size_t)blockIdx.x * SCAN_CHUNK + tid * 16;
#pragma unroll
  for (int i = 0; i < 16; ++i) atomicAdd(&cnt[p[i]], 1);
  __syncthreads();
  if (tid < NEXP) hist[blockIdx.x * NEXP + tid] = cnt[tid];
}

// ---------------------------------------------------------------------------
// Kernel 3: cross-chunk exclusive scan per expert + counters/dropped outputs.
// ---------------------------------------------------------------------------
__global__ __launch_bounds__(64) void scan_kernel(
    const int* __restrict__ hist, int* __restrict__ boff,
    float* __restrict__ out_cnt, float* __restrict__ out_drop, int nblocks)
{
  __shared__ float drop_s[NEXP];
  const int e = threadIdx.x;
  int run = 0;
  for (int b = 0; b < nblocks; ++b) {
    boff[b * NEXP + e] = run;
    run += hist[b * NEXP + e];
  }
  out_cnt[e] = (float)(run < CAPACITY ? run : CAPACITY);
  drop_s[e]  = (float)(run > CAPACITY ? run - CAPACITY : 0);
  __syncthreads();
  if (e == 0) {
    float d = 0.0f;
    for (int i = 0; i < NEXP; ++i) d += drop_s[i];
    *out_drop = d;
  }
}

// ---------------------------------------------------------------------------
// Kernel 4: exact in-order rank per assignment -> mask, renormalized weights.
// Thread t owns 16 consecutive assignments (8 whole tokens).
// ---------------------------------------------------------------------------
__global__ __launch_bounds__(256) void rank_kernel(
    const unsigned char* __restrict__ eid, const float* __restrict__ pw,
    const int* __restrict__ boff,
    float* __restrict__ out_w, float* __restrict__ out_m)
{
  __shared__ unsigned short pre[NEXP * 257];     // [e][t] counts -> excl prefix
  const int tid = threadIdx.x;
  for (int i = tid; i < NEXP * 257; i += 256) pre[i] = 0;
  __syncthreads();

  const int f0 = blockIdx.x * SCAN_CHUNK + tid * 16;
  unsigned char e[16];
#pragma unroll
  for (int i = 0; i < 16; ++i) e[i] = eid[f0 + i];
#pragma unroll
  for (int i = 0; i < 16; ++i) pre[e[i] * 257 + tid] += 1;  // column owned by t
  __syncthreads();

  if (tid < NEXP) {                               // serial excl scan over threads
    unsigned short accp = 0;
    unsigned short* row = &pre[tid * 257];
    for (int t = 0; t < 256; ++t) { unsigned short c = row[t]; row[t] = accp; accp += c; }
  }
  __syncthreads();

  float mask[16];
#pragma unroll
  for (int i = 0; i < 16; ++i) {
    int rl = 0;
#pragma unroll
    for (int j = 0; j < 16; ++j) if (j < i) rl += (e[j] == e[i]);
    const int g = boff[blockIdx.x * NEXP + e[i]] + (int)pre[e[i] * 257 + tid] + rl;
    mask[i] = (g < CAPACITY) ? 1.0f : 0.0f;
  }

#pragma unroll
  for (int j = 0; j < 8; ++j) {                   // finalize 8 tokens
    const int f = f0 + 2 * j;
    const float m1 = mask[2 * j], m2 = mask[2 * j + 1];
    const float inv = 1.0f / (m1 + m2 + 1e-10f);
    out_w[f + 0] = pw[f + 0] * m1 * inv;
    out_w[f + 1] = pw[f + 1] * m2 * inv;
    out_m[f + 0] = m1;
    out_m[f + 1] = m2;
  }
}

// ---------------------------------------------------------------------------
extern "C" void kernel_launch(void* const* d_in, const int* in_sizes, int n_in,
                              void* d_out, int out_size, void* d_ws, size_t ws_size,
                              hipStream_t stream) {
  const float* x = (const float*)d_in[0];
  const float* W = (const float*)d_in[1];
  const int N = in_sizes[0] / HIDDEN;             // 262144

  float* out      = (float*)d_out;
  float* out_idx  = out;                          // [2N] indices (as float)
  float* out_w    = out + (size_t)2 * N;          // [2N] weights
  float* out_m    = out + (size_t)4 * N;          // [2N] capacity mask
  float* out_cnt  = out + (size_t)6 * N;          // [64] expert counters
  float* out_drop = out + (size_t)6 * N + NEXP;   // [1]  num_dropped

  char* ws = (char*)d_ws;
  unsigned char* eid = (unsigned char*)ws;                    // 2N bytes
  float* pw  = (float*)(ws + (size_t)2 * N);                  // 2N floats
  int*   hist = (int*)(ws + (size_t)2 * N + (size_t)8 * N);   // [nscan*64]
  const int nscan = (2 * N) / SCAN_CHUNK;                     // 128
  int*   boff = hist + (size_t)nscan * NEXP;                  // [nscan*64]
  unsigned short* Wt = (unsigned short*)(boff + (size_t)nscan * NEXP); // 128KB

  convert_W<<<NEXP, 256, 0, stream>>>(W, Wt);
  router_gemm_topk<<<N / TPB_TOK, 256, 0, stream>>>(x, Wt, out_idx, eid, pw);
  hist_kernel<<<nscan, 256, 0, stream>>>(eid, hist);
  scan_kernel<<<1, 64, 0, stream>>>(hist, boff, out_cnt, out_drop, nscan);
  rank_kernel<<<nscan, 256, 0, stream>>>(eid, pw, boff, out_w, out_m);
}
